// GraphTransformerLayer_67095979098897
// MI455X (gfx1250) — compile-verified, matrix operand order checked by reading, not run on hardware
//
#include <hip/hip_runtime.h>
#include <hip/hip_bf16.h>

// ---------------------------------------------------------------------------
// Graph transformer layer for MI455X (gfx1250, wave32, WMMA).
// N=2048 tokens, D=1024 model dim, H=16 heads, HD=64, DFF=4096.
// All GEMMs + attention use v_wmma_f32_16x16x32_bf16 (fp32 accumulate).
// GEMM uses double-buffered async global->LDS copies (ASYNCcnt pipeline),
// falling back to synchronous copies if the builtins are unavailable.
// ---------------------------------------------------------------------------

#define NTOK 2048
#define DMODEL 1024
#define NHEAD 16
#define HDIM 64
#define DFF 4096

typedef __bf16 bf16_t;
typedef __attribute__((ext_vector_type(16))) __bf16 v16bf;
typedef __attribute__((ext_vector_type(8)))  __bf16 v8bf;
typedef __attribute__((ext_vector_type(8)))  float  v8f;
typedef __attribute__((ext_vector_type(4)))  int    v4i;

#ifndef __has_builtin
#define __has_builtin(x) 0
#endif

#if __has_builtin(__builtin_amdgcn_global_load_async_to_lds_b128)
#define ASYNC_LDS 1
typedef __attribute__((address_space(1))) v4i* gas_v4i;
typedef __attribute__((address_space(3))) v4i* las_v4i;
#else
#define ASYNC_LDS 0
#endif

// 16-byte global -> LDS copy (async on gfx1250 when available)
__device__ __forceinline__ void copy16_g2l(const bf16_t* __restrict__ g,
                                           bf16_t* __restrict__ l) {
#if ASYNC_LDS
    __builtin_amdgcn_global_load_async_to_lds_b128((gas_v4i)g, (las_v4i)l, 0, 0);
#else
    *(v8bf*)l = *(const v8bf*)g;
#endif
}

__device__ __forceinline__ void wait_async_le4() {
#if ASYNC_LDS
#if __has_builtin(__builtin_amdgcn_s_wait_asynccnt)
    __builtin_amdgcn_s_wait_asynccnt(4);
#else
    asm volatile("s_wait_asynccnt 4" ::: "memory");
#endif
#endif
}
__device__ __forceinline__ void wait_async_le0() {
#if ASYNC_LDS
#if __has_builtin(__builtin_amdgcn_s_wait_asynccnt)
    __builtin_amdgcn_s_wait_asynccnt(0);
#else
    asm volatile("s_wait_asynccnt 0" ::: "memory");
#endif
#endif
}

// Load one 16x32 bf16 A/B fragment (wave32 layout per CDNA5 ISA 7.12.2):
//   lanes 0-15  : row r = lane,    elems 0..7 -> K 0..7,   elems 8..15 -> K 16..23
//   lanes 16-31 : row r = lane-16, elems 0..7 -> K 8..15,  elems 8..15 -> K 24..31
// p points at (row0, k0) of a row-major [16][>=32] region with leading dim ld.
__device__ __forceinline__ v16bf load_frag(const bf16_t* __restrict__ p, int ld) {
    const int lane = threadIdx.x & 31;
    const int r    = lane & 15;
    const int kh   = lane >> 4;
    const bf16_t* q = p + (size_t)r * ld + kh * 8;
    v8bf lo = *(const v8bf*)(q);
    v8bf hi = *(const v8bf*)(q + 16);
    return __builtin_shufflevector(lo, hi, 0,1,2,3,4,5,6,7,8,9,10,11,12,13,14,15);
}

__device__ __forceinline__ v8f wmma_bf16(v16bf a, v16bf b, v8f c) {
    return __builtin_amdgcn_wmma_f32_16x16x32_bf16(false, a, false, b, (short)0, c,
                                                   false, false);
}

// ---------------------------------------------------------------------------
// LayerNorm (fp32 in) -> bf16 out, one block (8 waves) per row of 1024.
// ---------------------------------------------------------------------------
__global__ __launch_bounds__(256) void ln_cast_kernel(
    const float* __restrict__ x, const float* __restrict__ g,
    const float* __restrict__ b, bf16_t* __restrict__ out, int D)
{
    const int row  = blockIdx.x;
    const int tid  = threadIdx.x;
    const int lane = tid & 31;
    const int wave = tid >> 5;
    const float* xr = x + (size_t)row * D;

    float s = 0.f, s2 = 0.f;
    for (int i = tid; i < D; i += 256) {
        float v = xr[i];
        s += v; s2 += v * v;
    }
    #pragma unroll
    for (int off = 16; off >= 1; off >>= 1) {
        s  += __shfl_xor(s,  off, 32);
        s2 += __shfl_xor(s2, off, 32);
    }
    __shared__ float rs[8], rs2[8];
    if (lane == 0) { rs[wave] = s; rs2[wave] = s2; }
    __syncthreads();
    float ts = 0.f, ts2 = 0.f;
    #pragma unroll
    for (int w = 0; w < 8; ++w) { ts += rs[w]; ts2 += rs2[w]; }
    const float mean = ts / (float)D;
    const float var  = ts2 / (float)D - mean * mean;
    const float rstd = rsqrtf(var + 1e-5f);

    bf16_t* orow = out + (size_t)row * D;
    for (int i = tid; i < D; i += 256) {
        float v = (xr[i] - mean) * rstd * g[i] + b[i];
        orow[i] = (bf16_t)v;
    }
}

// ---------------------------------------------------------------------------
// W [K][Nn] fp32 -> Wt [Nn][K] bf16 (fragment-friendly: B rows contiguous in K)
// block (32,8), grid (Nn/32, K/32)
// ---------------------------------------------------------------------------
__global__ __launch_bounds__(256) void transpose_cast_kernel(
    const float* __restrict__ W, bf16_t* __restrict__ Wt, int K, int Nn)
{
    __shared__ float t[32][33];
    const int n0 = blockIdx.x * 32;
    const int k0 = blockIdx.y * 32;
    #pragma unroll
    for (int dy = 0; dy < 32; dy += 8)
        t[threadIdx.y + dy][threadIdx.x] =
            W[(size_t)(k0 + threadIdx.y + dy) * Nn + n0 + threadIdx.x];
    __syncthreads();
    #pragma unroll
    for (int dy = 0; dy < 32; dy += 8)
        Wt[(size_t)(n0 + threadIdx.y + dy) * K + k0 + threadIdx.x] =
            (bf16_t)t[threadIdx.x][threadIdx.y + dy];
}

// ---------------------------------------------------------------------------
// GEMM: out = act(A[M][K](bf16) @ Bt[N][K]^T(bf16) + bias) (+ res)
// Block tile 128x128, 8 waves each 32x64 (2x4 WMMA tiles), K chunk 32.
// Double-buffered LDS fed by async global->LDS copies (ASYNCcnt pipeline).
// Optional fp32 / bf16 / transposed-bf16 stores.
// ---------------------------------------------------------------------------
__global__ __launch_bounds__(256) void gemm_bf16_kernel(
    const bf16_t* __restrict__ A, const bf16_t* __restrict__ Bt,
    const float* __restrict__ bias, const float* __restrict__ res,
    float* __restrict__ outf, bf16_t* __restrict__ outb,
    bf16_t* __restrict__ outbT, int M, int Nn, int K, int gelu)
{
    __shared__ __attribute__((aligned(16))) bf16_t As[2][128][40];
    __shared__ __attribute__((aligned(16))) bf16_t Bs[2][128][40];

    const int tid  = threadIdx.x;
    const int lane = tid & 31;
    const int wave = tid >> 5;
    const int half = lane >> 4;
    const int lcol = lane & 15;
    const int wm   = wave & 3;       // 0..3 -> 32-row slice
    const int wn   = wave >> 2;      // 0..1 -> 64-col slice
    const int m0   = blockIdx.y * 128;
    const int n0   = blockIdx.x * 128;

    // cooperative tile loads: 128 rows x 32 bf16 each for A and B,
    // thread t copies 2x16B of A and 2x16B of B per K-chunk.
    const int crow = tid >> 1, chalf = tid & 1;
    const bf16_t* aBase = A  + (size_t)(m0 + crow) * K + chalf * 16;
    const bf16_t* bBase = Bt + (size_t)(n0 + crow) * K + chalf * 16;

    v8f acc[2][4];
    #pragma unroll
    for (int i = 0; i < 2; ++i)
        #pragma unroll
        for (int j = 0; j < 4; ++j)
            acc[i][j] = (v8f){0.f,0.f,0.f,0.f,0.f,0.f,0.f,0.f};

    const int nk = K >> 5;

    // prologue: issue chunk 0 into buffer 0
    {
        bf16_t* al = &As[0][crow][chalf * 16];
        bf16_t* bl = &Bs[0][crow][chalf * 16];
        copy16_g2l(aBase,     al);
        copy16_g2l(aBase + 8, al + 8);
        copy16_g2l(bBase,     bl);
        copy16_g2l(bBase + 8, bl + 8);
    }

    int buf = 0;
    for (int kc = 0; kc < nk; ++kc) {
        if (kc + 1 < nk) {
            const int koff = (kc + 1) << 5;
            bf16_t* al = &As[buf ^ 1][crow][chalf * 16];
            bf16_t* bl = &Bs[buf ^ 1][crow][chalf * 16];
            copy16_g2l(aBase + koff,     al);
            copy16_g2l(aBase + koff + 8, al + 8);
            copy16_g2l(bBase + koff,     bl);
            copy16_g2l(bBase + koff + 8, bl + 8);
            wait_async_le4();     // chunk kc's 4 async ops complete (in-order)
        } else {
            wait_async_le0();
        }
        __syncthreads();          // all waves' chunk kc data visible in LDS

        v16bf a0 = load_frag(&As[buf][wm * 32][0],      40);
        v16bf a1 = load_frag(&As[buf][wm * 32 + 16][0], 40);
        #pragma unroll
        for (int j = 0; j < 4; ++j) {
            v16bf bfrag = load_frag(&Bs[buf][wn * 64 + j * 16][0], 40);
            acc[0][j] = wmma_bf16(a0, bfrag, acc[0][j]);
            acc[1][j] = wmma_bf16(a1, bfrag, acc[1][j]);
        }
        __syncthreads();          // buffer may be overwritten next iteration
        buf ^= 1;
    }

    #pragma unroll
    for (int mi = 0; mi < 2; ++mi) {
        #pragma unroll
        for (int ni = 0; ni < 4; ++ni) {
            const int col = n0 + wn * 64 + ni * 16 + lcol;
            const float bv = bias ? bias[col] : 0.f;
            #pragma unroll
            for (int r = 0; r < 8; ++r) {
                const int row = m0 + wm * 32 + mi * 16 + r + 8 * half;
                float v = acc[mi][ni][r] + bv;
                if (gelu) v = 0.5f * v * (1.f + erff(v * 0.70710678118654752f));
                if (res)  v += res[(size_t)row * Nn + col];
                if (outf)  outf[(size_t)row * Nn + col] = v;
                if (outb)  outb[(size_t)row * Nn + col] = (bf16_t)v;
                if (outbT) outbT[(size_t)col * M + row] = (bf16_t)v;
            }
        }
    }
}

// ---------------------------------------------------------------------------
// Flash attention, one head per blockIdx.x, 64 query rows per blockIdx.y.
// 4 waves; wave owns 16 query rows. Online softmax over 32 key blocks of 64.
// S = Q Kt /8 + spatial_bias ; O = softmax(S) V ; Vt is [D][N] (pre-transposed).
// grid = (H, N/64) so the 16 head-blocks of a query block share bias in L2.
// ---------------------------------------------------------------------------
__global__ __launch_bounds__(128) void flash_attn_kernel(
    const bf16_t* __restrict__ Qb, const bf16_t* __restrict__ Kb,
    const bf16_t* __restrict__ Vt, const float* __restrict__ sbias,
    bf16_t* __restrict__ Ob)
{
    __shared__ __attribute__((aligned(16))) bf16_t Ps[4][16][80];

    const int tid  = threadIdx.x;
    const int lane = tid & 31;
    const int wave = tid >> 5;
    const int half = lane >> 4;
    const int lcol = lane & 15;
    const int hc   = blockIdx.x * HDIM;
    const int qb0  = blockIdx.y * 64;
    const int q0   = qb0 + wave * 16;

    v16bf aq[2];
    aq[0] = load_frag(Qb + (size_t)q0 * DMODEL + hc,      DMODEL);
    aq[1] = load_frag(Qb + (size_t)q0 * DMODEL + hc + 32, DMODEL);

    v8f o[4], m, l;
    #pragma unroll
    for (int nt = 0; nt < 4; ++nt) o[nt] = (v8f){0.f,0.f,0.f,0.f,0.f,0.f,0.f,0.f};
    #pragma unroll
    for (int r = 0; r < 8; ++r) { m[r] = -3.0e38f; l[r] = 0.f; }

    for (int jb = 0; jb < NTOK; jb += 64) {
        // prefetch next key block rows + next spatial-bias columns
        const int jn = jb + 64;
        if (jn < NTOK) {
            if (tid < 64) {
                __builtin_prefetch(Kb + (size_t)(jn + tid) * DMODEL + hc, 0, 1);
            } else {
                __builtin_prefetch(sbias + (size_t)(qb0 + tid - 64) * NTOK + jn, 0, 1);
            }
        }

        v8f s[4];
        #pragma unroll
        for (int nt = 0; nt < 4; ++nt) {
            v8f a = (v8f){0.f,0.f,0.f,0.f,0.f,0.f,0.f,0.f};
            #pragma unroll
            for (int kc = 0; kc < 2; ++kc) {
                v16bf bk = load_frag(
                    Kb + (size_t)(jb + nt * 16) * DMODEL + hc + kc * 32, DMODEL);
                a = wmma_bf16(aq[kc], bk, a);
            }
            s[nt] = a;
        }
        // scale + spatial bias (bias shared by all heads -> L2 resident)
        #pragma unroll
        for (int nt = 0; nt < 4; ++nt) {
            const int col = jb + nt * 16 + lcol;
            #pragma unroll
            for (int r = 0; r < 8; ++r) {
                const int row = q0 + r + 8 * half;
                s[nt][r] = s[nt][r] * 0.125f + sbias[(size_t)row * NTOK + col];
            }
        }
        // online softmax over the key axis (rows live in VGPR index r)
        v8f rmax = s[0];
        #pragma unroll
        for (int nt = 1; nt < 4; ++nt)
            #pragma unroll
            for (int r = 0; r < 8; ++r) rmax[r] = fmaxf(rmax[r], s[nt][r]);
        #pragma unroll
        for (int off = 1; off <= 8; off <<= 1)
            #pragma unroll
            for (int r = 0; r < 8; ++r)
                rmax[r] = fmaxf(rmax[r], __shfl_xor(rmax[r], off, 32));

        v8f mnew, scale, rsum;
        #pragma unroll
        for (int r = 0; r < 8; ++r) {
            mnew[r]  = fmaxf(m[r], rmax[r]);
            scale[r] = __expf(m[r] - mnew[r]);
            rsum[r]  = 0.f;
        }
        #pragma unroll
        for (int nt = 0; nt < 4; ++nt)
            #pragma unroll
            for (int r = 0; r < 8; ++r) {
                float p = __expf(s[nt][r] - mnew[r]);
                s[nt][r] = p;
                rsum[r] += p;
            }
        #pragma unroll
        for (int off = 1; off <= 8; off <<= 1)
            #pragma unroll
            for (int r = 0; r < 8; ++r) rsum[r] += __shfl_xor(rsum[r], off, 32);
        #pragma unroll
        for (int r = 0; r < 8; ++r) {
            l[r] = l[r] * scale[r] + rsum[r];
            m[r] = mnew[r];
        }
        #pragma unroll
        for (int nt = 0; nt < 4; ++nt)
            #pragma unroll
            for (int r = 0; r < 8; ++r) o[nt][r] *= scale[r];

        // C-layout P -> A-layout bf16 fragments via LDS
        __syncthreads();
        #pragma unroll
        for (int nt = 0; nt < 4; ++nt)
            #pragma unroll
            for (int r = 0; r < 8; ++r)
                Ps[wave][r + 8 * half][nt * 16 + lcol] = (bf16_t)s[nt][r];
        __syncthreads();

        #pragma unroll
        for (int kc = 0; kc < 2; ++kc) {
            v16bf ap = load_frag(&Ps[wave][0][0] + kc * 32, 80);
            #pragma unroll
            for (int nt = 0; nt < 4; ++nt) {
                v16bf bv = load_frag(
                    Vt + (size_t)(hc + nt * 16) * NTOK + jb + kc * 32, NTOK);
                o[nt] = wmma_bf16(ap, bv, o[nt]);
            }
        }
    }

    #pragma unroll
    for (int nt = 0; nt < 4; ++nt) {
        const int col = hc + nt * 16 + lcol;
        #pragma unroll
        for (int r = 0; r < 8; ++r) {
            const int row = q0 + r + 8 * half;
            Ob[(size_t)row * DMODEL + col] = (bf16_t)(o[nt][r] / l[r]);
        }
    }
}

// ---------------------------------------------------------------------------
extern "C" void kernel_launch(void* const* d_in, const int* in_sizes, int n_in,
                              void* d_out, int out_size, void* d_ws, size_t ws_size,
                              hipStream_t stream)
{
    (void)in_sizes; (void)n_in; (void)out_size; (void)ws_size;

    const float* h   = (const float*)d_in[0];
    const float* sb  = (const float*)d_in[1];
    const float* Wq  = (const float*)d_in[2];
    const float* bq  = (const float*)d_in[3];
    const float* Wk  = (const float*)d_in[4];
    const float* bk  = (const float*)d_in[5];
    const float* Wv  = (const float*)d_in[6];
    const float* bv  = (const float*)d_in[7];
    const float* Wo  = (const float*)d_in[8];
    const float* bo  = (const float*)d_in[9];
    const float* W1  = (const float*)d_in[10];
    const float* b1  = (const float*)d_in[11];
    const float* W2  = (const float*)d_in[12];
    const float* b2  = (const float*)d_in[13];
    const float* g1  = (const float*)d_in[14];
    const float* be1 = (const float*)d_in[15];
    const float* g2  = (const float*)d_in[16];
    const float* be2 = (const float*)d_in[17];
    float* out = (float*)d_out;

    char* base = (char*)d_ws;
    size_t off = 0;
    auto take = [&](size_t nbytes) -> void* {
        void* p = base + off;
        off += (nbytes + 255) & ~(size_t)255;
        return p;
    };
    bf16_t* WqT  = (bf16_t*)take((size_t)DMODEL * DMODEL * 2);
    bf16_t* WkT  = (bf16_t*)take((size_t)DMODEL * DMODEL * 2);
    bf16_t* WvT  = (bf16_t*)take((size_t)DMODEL * DMODEL * 2);
    bf16_t* WoT  = (bf16_t*)take((size_t)DMODEL * DMODEL * 2);
    bf16_t* W1T  = (bf16_t*)take((size_t)DFF * DMODEL * 2);
    bf16_t* W2T  = (bf16_t*)take((size_t)DMODEL * DFF * 2);
    bf16_t* hn   = (bf16_t*)take((size_t)NTOK * DMODEL * 2);
    bf16_t* Qb   = (bf16_t*)take((size_t)NTOK * DMODEL * 2);
    bf16_t* Kb   = (bf16_t*)take((size_t)NTOK * DMODEL * 2);
    bf16_t* VtT  = (bf16_t*)take((size_t)DMODEL * NTOK * 2);
    bf16_t* attn = (bf16_t*)take((size_t)NTOK * DMODEL * 2);
    float*  h2   = (float*) take((size_t)NTOK * DMODEL * 4);
    bf16_t* hn2  = (bf16_t*)take((size_t)NTOK * DMODEL * 2);
    bf16_t* ffn1 = (bf16_t*)take((size_t)NTOK * DFF * 2);

    const dim3 tb(32, 8);
    transpose_cast_kernel<<<dim3(DMODEL/32, DMODEL/32), tb, 0, stream>>>(Wq, WqT, DMODEL, DMODEL);
    transpose_cast_kernel<<<dim3(DMODEL/32, DMODEL/32), tb, 0, stream>>>(Wk, WkT, DMODEL, DMODEL);
    transpose_cast_kernel<<<dim3(DMODEL/32, DMODEL/32), tb, 0, stream>>>(Wv, WvT, DMODEL, DMODEL);
    transpose_cast_kernel<<<dim3(DMODEL/32, DMODEL/32), tb, 0, stream>>>(Wo, WoT, DMODEL, DMODEL);
    transpose_cast_kernel<<<dim3(DFF/32,    DMODEL/32), tb, 0, stream>>>(W1, W1T, DMODEL, DFF);
    transpose_cast_kernel<<<dim3(DMODEL/32, DFF/32),    tb, 0, stream>>>(W2, W2T, DFF, DMODEL);

    // LN1 -> bf16
    ln_cast_kernel<<<NTOK, 256, 0, stream>>>(h, g1, be1, hn, DMODEL);

    // Q, K row-major bf16; V stored transposed ([d][n]) for the PV matmul
    gemm_bf16_kernel<<<dim3(DMODEL/128, NTOK/128), 256, 0, stream>>>(
        hn, WqT, bq, nullptr, nullptr, Qb, nullptr, NTOK, DMODEL, DMODEL, 0);
    gemm_bf16_kernel<<<dim3(DMODEL/128, NTOK/128), 256, 0, stream>>>(
        hn, WkT, bk, nullptr, nullptr, Kb, nullptr, NTOK, DMODEL, DMODEL, 0);
    gemm_bf16_kernel<<<dim3(DMODEL/128, NTOK/128), 256, 0, stream>>>(
        hn, WvT, bv, nullptr, nullptr, nullptr, VtT, NTOK, DMODEL, DMODEL, 0);

    // Fused attention (softmax over keys, spatial bias, per head)
    flash_attn_kernel<<<dim3(NHEAD, NTOK/64), 128, 0, stream>>>(Qb, Kb, VtT, sb, attn);

    // Output projection + residual -> h2 (fp32)
    gemm_bf16_kernel<<<dim3(DMODEL/128, NTOK/128), 256, 0, stream>>>(
        attn, WoT, bo, h, h2, nullptr, nullptr, NTOK, DMODEL, DMODEL, 0);

    // LN2 -> bf16
    ln_cast_kernel<<<NTOK, 256, 0, stream>>>(h2, g2, be2, hn2, DMODEL);

    // FFN: gelu(hn2 @ W1 + b1) -> bf16 ; @ W2 + b2 + h2 -> d_out (fp32)
    gemm_bf16_kernel<<<dim3(DFF/128, NTOK/128), 256, 0, stream>>>(
        hn2, W1T, b1, nullptr, nullptr, ffn1, nullptr, NTOK, DFF, DMODEL, 1);
    gemm_bf16_kernel<<<dim3(DMODEL/128, NTOK/128), 256, 0, stream>>>(
        ffn1, W2T, b2, h2, out, nullptr, nullptr, NTOK, DMODEL, DFF, 0);
}
